// PhysicsRegularizedAttention_19928648253542
// MI455X (gfx1250) — matrix operand order repaired and checked by reading, not compiled, not run on hardware
//
#include <hip/hip_runtime.h>
#include <hip/hip_bf16.h>

typedef __attribute__((ext_vector_type(16))) _Float16 v16h;
typedef __attribute__((ext_vector_type(8)))  float    v8f;
typedef __attribute__((ext_vector_type(4)))  float    f32x4;

// ---------------- WMMA helpers (CDNA5 wave32, 16x16x32 f16 -> f32) ----------------

__device__ __forceinline__ v8f wmma_f16(v16h a, v16h b, v8f c) {
    return __builtin_amdgcn_wmma_f32_16x16x32_f16(false, a, false, b, (short)0, c, false, false);
}

// A fragment: 16(M) x 32(K) f16 from row-major [.., ld] source.
// lane<16: row M=lane, k = {0..7, 16..23}; lane>=16: row M=lane-16, k = {8..15, 24..31}
__device__ __forceinline__ v16h ldA(const _Float16* p, int ld, int lane) {
    const _Float16* q = p + (size_t)(lane & 15) * ld + ((lane >> 4) << 3);
    v16h a;
    ((f32x4*)&a)[0] = *(const f32x4*)(q);
    ((f32x4*)&a)[1] = *(const f32x4*)(q + 16);
    return a;
}

// B fragment: 32(K) x 16(N). Source stored row-major as [N][K] (i.e. column n of B is row n).
// lane<16: col N=lane, k = {0..7, 16..23}; lane>=16: col N=lane-16, k = {8..15, 24..31}
__device__ __forceinline__ v16h ldB(const _Float16* p, int ld, int lane) {
    const _Float16* q = p + (size_t)(lane & 15) * ld + ((lane >> 4) << 3);
    v16h b;
    ((f32x4*)&b)[0] = *(const f32x4*)(q);
    ((f32x4*)&b)[1] = *(const f32x4*)(q + 16);
    return b;
}

// ---------------- constants ----------------
#define B_   8
#define N_   1024
#define C_   256
#define NH_  4
#define HD_  64
#define SLD  1040   // LDS row stride (floats) for 32x1024 score strip, padded vs 64 banks

// ---------------- k0: fp32 -> f16 conversions ----------------
__global__ void cvt_kernel(const float* __restrict__ x, _Float16* __restrict__ xh, int n1,
                           const float* __restrict__ w1, _Float16* __restrict__ w1h, int n2,
                           const float* __restrict__ w2, _Float16* __restrict__ w2h, int n3) {
    int i = blockIdx.x * blockDim.x + threadIdx.x;
    if (i < n1) xh[i]  = (_Float16)x[i];
    if (i < n2) w1h[i] = (_Float16)w1[i];
    if (i < n3) w2h[i] = (_Float16)w2[i];
}

// ---------------- k1: physics prior map: sigmoid(conv3x3(mean_C(x), mean_k(kernels))) ----------------
__global__ void amap_kernel(const float* __restrict__ x, const float* __restrict__ kernels,
                            float* __restrict__ amap) {
    __shared__ float xm[N_];
    int b = blockIdx.x;
    int t = threadIdx.x;                 // 1024 threads = one spatial position each
    const float* xp = x + ((size_t)b * N_ + t) * C_;
    float s = 0.f;
    for (int c = 0; c < C_; ++c) s += xp[c];
    xm[t] = s * (1.0f / C_);
    __syncthreads();
    float km[9];
    for (int i = 0; i < 9; ++i)
        km[i] = 0.25f * (kernels[i] + kernels[9 + i] + kernels[18 + i] + kernels[27 + i]);
    int i = t >> 5, j = t & 31;          // H=W=32
    float acc = 0.f;
    for (int di = -1; di <= 1; ++di)
        for (int dj = -1; dj <= 1; ++dj) {
            int ii = i + di, jj = j + dj;
            if (ii >= 0 && ii < 32 && jj >= 0 && jj < 32)
                acc += km[(di + 1) * 3 + (dj + 1)] * xm[ii * 32 + jj];
        }
    amap[b * N_ + t] = 1.0f / (1.0f + __expf(-acc));
}

// ---------------- k2: regularization loss (tiny) ----------------
__global__ void reg_kernel(const float* __restrict__ kernels, float* __restrict__ outloss) {
    // CONSTRAINTS: ARC, CURV3 (analytic bilinear 5->3 of CURV5), RAD, TAN
    const float Cst[36] = {
        -1.f, -1.f, -1.f,  2.f, 2.f, 2.f,  -1.f, -1.f, -1.f,
        -1.f/9.f, 0.f, -1.f/9.f,  0.f, 4.f, 0.f,  -1.f/9.f, 0.f, -1.f/9.f,
         0.f, -1.f, 0.f,  -1.f, 4.f, -1.f,  0.f, -1.f, 0.f,
        -1.f, 0.f, -1.f,  0.f, 4.f, 0.f,  -1.f, 0.f, -1.f };
    float mse = 0.f, l1 = 0.f;
    for (int i = 0; i < 36; ++i) {
        float d = kernels[i] - Cst[i];
        mse += d * d;
        l1  += fabsf(kernels[i]);
    }
    *outloss = mse * (1.0f / 9.0f) + 0.01f * l1;
}

// ---------------- k3: QKV = Xh @ Wqkv^T + b  (WMMA), scatter to q, k, vT (f16) ----------------
__global__ __launch_bounds__(256) void qkv_kernel(const _Float16* __restrict__ xh,
                                                  const _Float16* __restrict__ wh,
                                                  const float* __restrict__ bias,
                                                  _Float16* __restrict__ q,
                                                  _Float16* __restrict__ k,
                                                  _Float16* __restrict__ vT) {
    int wave = (blockIdx.x * blockDim.x + threadIdx.x) >> 5;   // 24576 tiles: 512 M x 48 J
    int lane = threadIdx.x & 31;
    int mt = wave / 48, jt = wave % 48;
    const _Float16* ap = xh + (size_t)mt * 16 * C_;
    const _Float16* bp = wh + (size_t)jt * 16 * C_;
    v8f acc = {};
    for (int kk = 0; kk < C_; kk += 32)
        acc = wmma_f16(ldA(ap + kk, C_, lane), ldB(bp + kk, C_, lane), acc);
    int n  = lane & 15;
    int mb = (lane >> 4) * 8;
    int j  = jt * 16 + n;
    float bj = bias[j];
    int which = j >> 8;            // 0=q 1=k 2=v
    int c = j & 255;
    int h = c >> 6, d = c & 63;
    for (int r = 0; r < 8; ++r) {
        int R  = mt * 16 + mb + r;
        int b_ = R >> 10, nn = R & 1023;
        _Float16 val = (_Float16)(acc[r] + bj);
        size_t bh = ((size_t)b_ * NH_ + h);
        if (which == 0)      q [(bh * N_ + nn) * HD_ + d] = val;
        else if (which == 1) k [(bh * N_ + nn) * HD_ + d] = val;
        else                 vT[(bh * HD_ + d) * N_ + nn] = val;   // V transposed
    }
}

// ---------------- k4: attention (per batch, 32-row strip; loops heads) ----------------
__global__ __launch_bounds__(256) void attn_kernel(const _Float16* __restrict__ q,
                                                   const _Float16* __restrict__ k,
                                                   const _Float16* __restrict__ vT,
                                                   const float* __restrict__ amap,
                                                   float* __restrict__ attnw,
                                                   _Float16* __restrict__ ctx) {
    extern __shared__ float S[];                 // [32][SLD] fp32 score strip (133 KB)
    int b    = blockIdx.x >> 5;
    int rs   = (blockIdx.x & 31) * 32;           // query-row strip base
    int wave = threadIdx.x >> 5;
    int lane = threadIdx.x & 31;
    int tid  = threadIdx.x;
    const float scale = 0.125f;                  // hd^-0.5
    const float* amb = amap + b * N_;

    for (int h = 0; h < NH_; ++h) {
        const _Float16* qh = q  + (((size_t)b * NH_ + h) * N_) * HD_;
        const _Float16* kh = k  + (((size_t)b * NH_ + h) * N_) * HD_;
        const _Float16* vh = vT + (((size_t)b * NH_ + h) * HD_) * N_;

        // ---- phase 1: S = (Q Kt)*scale + 0.1*amap(row)*amap(col), 2x64 tiles ----
        for (int t = wave; t < 128; t += 8) {
            int mt = t & 1, nt = t >> 1;
            v8f acc = {};
            const _Float16* ap = qh + (size_t)(rs + mt * 16) * HD_;
            const _Float16* bp = kh + (size_t)(nt * 16) * HD_;
            for (int kk = 0; kk < HD_; kk += 32)
                acc = wmma_f16(ldA(ap + kk, HD_, lane), ldB(bp + kk, HD_, lane), acc);
            int n = lane & 15, mb = (lane >> 4) * 8;
            float amc = amb[nt * 16 + n];
            for (int r = 0; r < 8; ++r) {
                int m = mt * 16 + mb + r;
                S[m * SLD + nt * 16 + n] = acc[r] * scale + 0.1f * amb[rs + m] * amc;
            }
        }
        __syncthreads();

        // ---- phase 2: row softmax (wave per row, lanes stride columns) ----
        for (int rr = wave; rr < 32; rr += 8) {
            float* row = S + rr * SLD;
            float mx = -1e30f;
            for (int jx = lane; jx < N_; jx += 32) mx = fmaxf(mx, row[jx]);
            for (int o = 16; o; o >>= 1) mx = fmaxf(mx, __shfl_xor(mx, o, 32));
            float sum = 0.f;
            for (int jx = lane; jx < N_; jx += 32) { float e = __expf(row[jx] - mx); row[jx] = e; sum += e; }
            for (int o = 16; o; o >>= 1) sum += __shfl_xor(sum, o, 32);
            float inv = 1.0f / sum;
            for (int jx = lane; jx < N_; jx += 32) row[jx] *= inv;
        }
        __syncthreads();

        // ---- phase 3a: accumulate head mean into attention_weights ----
        float* aw = attnw + ((size_t)b * N_ + rs) * N_;
        for (int idx = tid; idx < 32 * N_; idx += 256) {
            int m = idx >> 10, n = idx & 1023;
            float p = 0.25f * S[m * SLD + n];
            if (h == 0) aw[(size_t)m * N_ + n]  = p;
            else        aw[(size_t)m * N_ + n] += p;
        }

        // ---- phase 3b: P @ V (A from LDS fp32->f16, B from vT), one 16x16 tile/wave ----
        {
            int mt = wave & 1, dt = wave >> 1;
            v8f acc = {};
            for (int kk = 0; kk < N_; kk += 32) {
                v16h a;
                {
                    int m  = mt * 16 + (lane & 15);
                    int ko = (lane >> 4) << 3;
                    const float* sp = S + m * SLD + kk + ko;
                    for (int i = 0; i < 8; ++i) a[i]     = (_Float16)sp[i];
                    for (int i = 0; i < 8; ++i) a[8 + i] = (_Float16)sp[16 + i];
                }
                v16h bf = ldB(vh + (size_t)(dt * 16) * N_ + kk, N_, lane);
                acc = wmma_f16(a, bf, acc);
            }
            int n = lane & 15, mb = (lane >> 4) * 8;
            for (int r = 0; r < 8; ++r) {
                int nglob = rs + mt * 16 + mb + r;
                ctx[((size_t)b * N_ + nglob) * C_ + h * HD_ + dt * 16 + n] = (_Float16)acc[r];
            }
        }
        __syncthreads();   // S reused by next head's phase 1
    }
}

// ---------------- k5: out = ctx @ proj_w^T + proj_b (WMMA, fp32 out) ----------------
__global__ __launch_bounds__(256) void proj_kernel(const _Float16* __restrict__ ctx,
                                                   const _Float16* __restrict__ wh,
                                                   const float* __restrict__ bias,
                                                   float* __restrict__ out) {
    int wave = (blockIdx.x * blockDim.x + threadIdx.x) >> 5;   // 8192 tiles: 512 M x 16 Co
    int lane = threadIdx.x & 31;
    int mt = wave >> 4, ct = wave & 15;
    const _Float16* ap = ctx + (size_t)mt * 16 * C_;
    const _Float16* bp = wh  + (size_t)ct * 16 * C_;
    v8f acc = {};
    for (int kk = 0; kk < C_; kk += 32)
        acc = wmma_f16(ldA(ap + kk, C_, lane), ldB(bp + kk, C_, lane), acc);
    int n = lane & 15, mb = (lane >> 4) * 8;
    float bj = bias[ct * 16 + n];
    for (int r = 0; r < 8; ++r) {
        int R = mt * 16 + mb + r;
        out[(size_t)R * C_ + ct * 16 + n] = acc[r] + bj;
    }
}

// ---------------- launcher ----------------
extern "C" void kernel_launch(void* const* d_in, const int* in_sizes, int n_in,
                              void* d_out, int out_size, void* d_ws, size_t ws_size,
                              hipStream_t stream) {
    const float* x      = (const float*)d_in[0];   // [8,1024,256]
    const float* qkv_w  = (const float*)d_in[1];   // [768,256]
    const float* qkv_b  = (const float*)d_in[2];   // [768]
    const float* proj_w = (const float*)d_in[3];   // [256,256]
    const float* proj_b = (const float*)d_in[4];   // [256]
    const float* kers   = (const float*)d_in[5];   // [4,3,3]
    // d_in[6]=H, d_in[7]=W (32,32 assumed)

    char* ws = (char*)d_ws;
    size_t off = 0;
    float*    amap   = (float*)(ws + off);    off += (size_t)B_ * N_ * 4;            // 32 KB
    _Float16* xh     = (_Float16*)(ws + off); off += (size_t)B_ * N_ * C_ * 2;       // 4 MB
    _Float16* qkvwh  = (_Float16*)(ws + off); off += (size_t)3 * C_ * C_ * 2;        // 384 KB
    _Float16* projwh = (_Float16*)(ws + off); off += (size_t)C_ * C_ * 2;            // 128 KB
    _Float16* qf     = (_Float16*)(ws + off); off += (size_t)B_ * NH_ * N_ * HD_ * 2;
    _Float16* kf     = (_Float16*)(ws + off); off += (size_t)B_ * NH_ * N_ * HD_ * 2;
    _Float16* vTf    = (_Float16*)(ws + off); off += (size_t)B_ * NH_ * HD_ * N_ * 2;
    _Float16* ctx    = (_Float16*)(ws + off); off += (size_t)B_ * N_ * C_ * 2;       // total ~21.5 MB

    float* out   = (float*)d_out;                          // [8,1024,256]
    float* attnw = out + (size_t)B_ * N_ * C_;             // [8,1024,1024]
    float* loss  = out + (size_t)out_size - 1;             // scalar

    int n1 = B_ * N_ * C_, n2 = 3 * C_ * C_, n3 = C_ * C_;
    cvt_kernel<<<(n1 + 255) / 256, 256, 0, stream>>>(x, xh, n1, qkv_w, qkvwh, n2, proj_w, projwh, n3);
    amap_kernel<<<B_, N_, 0, stream>>>(x, kers, amap);
    reg_kernel<<<1, 1, 0, stream>>>(kers, loss);
    qkv_kernel<<<3072, 256, 0, stream>>>(xh, qkvwh, qkv_b, qf, kf, vTf);
    attn_kernel<<<B_ * 32, 256, 32 * SLD * sizeof(float), stream>>>(qf, kf, vTf, amap, attnw, ctx);
    proj_kernel<<<1024, 256, 0, stream>>>(ctx, projwh, proj_b, out);
}
